// BiLSTM_CRF_46385646796998
// MI455X (gfx1250) — matrix operand order repaired and checked by reading, not compiled, not run on hardware
//
#include <hip/hip_runtime.h>
#include <math.h>

// ---------------------------------------------------------------------------
// BiLSTM-CRF for MI455X (gfx1250, wave32, WMMA f16->f32)
// ---------------------------------------------------------------------------
typedef __attribute__((ext_vector_type(8)))  _Float16 v8h;
typedef __attribute__((ext_vector_type(16))) _Float16 v16h;
typedef __attribute__((ext_vector_type(8)))  float    v8f;

#define N_TOK   16384   // S*B tokens, flat (S,B) order
#define EDIM    300
#define EPAD    320     // pad K to multiple of 32 for 16x16x32 WMMA
#define H2U     256
#define G4      1024    // 4*H2
#define SEQ     256
#define BATCH   64
#define NTAG    10
#define START_TAG 8
#define STOP_TAG  9

__device__ __forceinline__ v16h cat16(v8h lo, v8h hi) {
  return __builtin_shufflevector(lo, hi, 0,1,2,3,4,5,6,7,8,9,10,11,12,13,14,15);
}
__device__ __forceinline__ v8f wmma16(v16h a, v16h b, v8f c) {
  // emits v_wmma_f32_16x16x32_f16
  return __builtin_amdgcn_wmma_f32_16x16x32_f16(false, a, false, b, (short)0, c,
                                                false, false);
}
__device__ __forceinline__ float sigm(float x) { return 1.0f / (1.0f + expf(-x)); }

// ---------------------------------------------------------------------------
// Phase 0: embedding gather + f32->f16, padded to EPAD
// X16[m][e] = emb[sentence_flat[m]][e]   (token m in (S,B) flat order == flat
// order of emb[sentence] memory, per the reference's raw reshape)
// ---------------------------------------------------------------------------
__global__ void k_embed_gather(const int* __restrict__ sent,
                               const float* __restrict__ emb,
                               _Float16* __restrict__ X) {
  int m = blockIdx.x;            // token 0..16383
  int e = threadIdx.x;           // 0..319
  int tok = sent[m];
  float v = (e < EDIM) ? emb[(size_t)tok * EDIM + e] : 0.0f;
  X[(size_t)m * EPAD + e] = (_Float16)v;
}

// generic f32 -> f16 row convert with column padding
__global__ void k_cvt_pad(const float* __restrict__ src, _Float16* __restrict__ dst,
                          int rows, int scols, int dcols) {
  int i = blockIdx.x * blockDim.x + threadIdx.x;
  if (i >= rows * dcols) return;
  int r = i / dcols, c = i % dcols;
  dst[i] = (c < scols) ? (_Float16)src[r * scols + c] : (_Float16)0.0f;
}

__global__ void k_add2(const float* __restrict__ a, const float* __restrict__ b,
                       float* __restrict__ o, int n) {
  int i = blockIdx.x * blockDim.x + threadIdx.x;
  if (i < n) o[i] = a[i] + b[i];
}

// ---------------------------------------------------------------------------
// Phase 1: G_in[dir][m][n] = X[m,:] . wih[dir][n,:] + bih[n] + bhh[n]
// WMMA GEMM: M=16384, N=1024, K=320.  One wave -> (16-row m-tile) x (4 n-tiles)
// ---------------------------------------------------------------------------
__global__ __launch_bounds__(256) void k_gin_gemm(const _Float16* __restrict__ X,
                                                  const _Float16* __restrict__ wih16,
                                                  const float* __restrict__ bsum,
                                                  float* __restrict__ Gin) {
  int gwave = (blockIdx.x * 256 + threadIdx.x) >> 5;   // 0..32767
  int lane  = threadIdx.x & 31;
  int dir   = gwave >> 14;                             // 16384 waves per dir
  int rem   = gwave & 16383;
  int mtile = rem >> 4;                                // 0..1023
  int n0    = (rem & 15) * 64;                         // 4 tiles of 16 cols

  const _Float16* W = wih16 + (size_t)dir * G4 * EPAD;
  float*          G = Gin   + (size_t)dir * N_TOK * G4;
  const float*   bs = bsum  + dir * G4;

  v8f acc[4] = {};
  int arow  = mtile * 16 + (lane & 15);
  int akoff = (lane < 16) ? 0 : 8;     // A 16-bit layout: k chunks {a, a+16}
  int bkoff = (lane < 16) ? 0 : 8;     // B 16-bit layout: k chunks {b, b+16}
  int ccol  = lane & 15;

#pragma unroll
  for (int kt = 0; kt < EPAD / 32; ++kt) {
    int kb = kt * 32;
    const _Float16* ap = X + (size_t)arow * EPAD + kb + akoff;
    v16h a = cat16(*(const v8h*)ap, *(const v8h*)(ap + 16));
#pragma unroll
    for (int j = 0; j < 4; ++j) {
      int n = n0 + j * 16 + ccol;      // column owned by this lane
      const _Float16* bp = W + (size_t)n * EPAD + kb + bkoff;
      v16h b = cat16(*(const v8h*)bp, *(const v8h*)(bp + 16));
      acc[j] = wmma16(a, b, acc[j]);
    }
  }
  int crow0 = mtile * 16 + ((lane >= 16) ? 8 : 0);
#pragma unroll
  for (int j = 0; j < 4; ++j) {
    int n = n0 + j * 16 + ccol;
    float bv = bs[n];
#pragma unroll
    for (int r = 0; r < 8; ++r)
      G[(size_t)(crow0 + r) * G4 + n] = acc[j][r] + bv;
  }
}

// ---------------------------------------------------------------------------
// Phase 2: recurrent scan. One 1024-thread workgroup per direction.
// Per step: gates(64x1024) = G_in[t] + h_{t-1}(64x256,f16 in LDS) @ whh^T.
// Wave w: m-tile = w%4, j-tiles {2*(w/4), 2*(w/4)+1}; owns all 4 gates of its
// cell tiles -> LSTM elementwise fully in registers, c-state resident in VGPRs.
// ---------------------------------------------------------------------------
__global__ __launch_bounds__(1024) void k_lstm_rec(const _Float16* __restrict__ whh16,
                                                   const float* __restrict__ Gin,
                                                   const float* __restrict__ h0,
                                                   const float* __restrict__ c0,
                                                   _Float16* __restrict__ hs) {
  extern __shared__ _Float16 sh[];          // 2 x (64*256) f16 ping-pong = 64KB
  int dir = blockIdx.x;                     // 0 = forward, 1 = backward
  const _Float16* W  = whh16 + (size_t)dir * G4 * H2U;
  const float*    G  = Gin   + (size_t)dir * N_TOK * G4;
  _Float16*       HS = hs    + (size_t)dir * N_TOK * H2U;

  int tid  = threadIdx.x;
  int lane = tid & 31;
  int wave = tid >> 5;
  int mtile = wave & 3;
  int jt0   = (wave >> 2) * 2;

  // init h buffer 0 from h0[dir] (f32 -> f16)
  for (int i = tid; i < BATCH * H2U; i += 1024)
    sh[i] = (_Float16)h0[dir * BATCH * H2U + i];

  // resident cell state in C-tile layout
  int rbase = mtile * 16 + ((lane >= 16) ? 8 : 0);
  int ccol  = lane & 15;
  v8f cst[2];
#pragma unroll
  for (int j = 0; j < 2; ++j) {
    int u = (jt0 + j) * 16 + ccol;
#pragma unroll
    for (int r = 0; r < 8; ++r)
      cst[j][r] = c0[dir * BATCH * H2U + (rbase + r) * H2U + u];
  }
  __syncthreads();

  int arow  = mtile * 16 + (lane & 15);
  int akoff = (lane < 16) ? 0 : 8;
  int bkoff = (lane < 16) ? 0 : 8;

  for (int t = 0; t < SEQ; ++t) {
    int tt = dir ? (SEQ - 1 - t) : t;                 // timestep being processed
    const _Float16* hbuf = sh + (t & 1) * (BATCH * H2U);
    _Float16*      hnext = sh + ((t + 1) & 1) * (BATCH * H2U);
    const float*      Gt = G + (size_t)tt * BATCH * G4;

#pragma unroll
    for (int j = 0; j < 2; ++j) {
      int jt = jt0 + j;
      v8f acc[4];
#pragma unroll
      for (int g = 0; g < 4; ++g) {
        int n = g * H2U + jt * 16 + ccol;
        v8f c;
#pragma unroll
        for (int r = 0; r < 8; ++r)                   // C init from G_in (L2-hot)
          c[r] = Gt[(size_t)(rbase + r) * G4 + n];
#pragma unroll
        for (int kt = 0; kt < H2U / 32; ++kt) {
          const _Float16* ap = hbuf + arow * H2U + kt * 32 + akoff;
          v16h a = cat16(*(const v8h*)ap, *(const v8h*)(ap + 16));
          const _Float16* bp = W + (size_t)n * H2U + kt * 32 + bkoff;
          v16h b = cat16(*(const v8h*)bp, *(const v8h*)(bp + 16));
          c = wmma16(a, b, c);
        }
        acc[g] = c;
      }
      // LSTM cell, all in registers: i=acc[0], f=acc[1], g=acc[2], o=acc[3]
      int u = jt * 16 + ccol;
#pragma unroll
      for (int r = 0; r < 8; ++r) {
        float ig = sigm(acc[0][r]);
        float fg = sigm(acc[1][r]);
        float gg = tanhf(acc[2][r]);
        float og = sigm(acc[3][r]);
        float cn = fg * cst[j][r] + ig * gg;
        cst[j][r] = cn;
        _Float16 hv = (_Float16)(og * tanhf(cn));
        hnext[(rbase + r) * H2U + u] = hv;                          // for step t+1
        HS[(size_t)(tt * BATCH + rbase + r) * H2U + u] = hv;        // for projection
      }
    }
    __syncthreads();   // hnext complete before next step reads it
  }
}

// ---------------------------------------------------------------------------
// Phase 3: feats[m][tag] = concat(h_f[m], h_b[m]) . W_out[tag,:] + b_out
// (raw reshape => feats flat token k maps to lstm token m=k)
// ---------------------------------------------------------------------------
__global__ __launch_bounds__(256) void k_proj(const _Float16* __restrict__ hs,
                                              const float* __restrict__ Wout,
                                              const float* __restrict__ bout,
                                              float* __restrict__ feats) {
  __shared__ float Wl[NTAG * 512];
  __shared__ float bl[NTAG];
  for (int i = threadIdx.x; i < NTAG * 512; i += 256) Wl[i] = Wout[i];
  if (threadIdx.x < NTAG) bl[threadIdx.x] = bout[threadIdx.x];
  __syncthreads();

  int m = blockIdx.x * 256 + threadIdx.x;
  const _Float16* hf = hs + (size_t)m * H2U;
  const _Float16* hb = hs + (size_t)N_TOK * H2U + (size_t)m * H2U;
  float acc[NTAG];
#pragma unroll
  for (int tg = 0; tg < NTAG; ++tg) acc[tg] = bl[tg];
  for (int u = 0; u < H2U; ++u) {
    float x = (float)hf[u];
#pragma unroll
    for (int tg = 0; tg < NTAG; ++tg) acc[tg] += x * Wl[tg * 512 + u];
  }
  for (int u = 0; u < H2U; ++u) {
    float x = (float)hb[u];
#pragma unroll
    for (int tg = 0; tg < NTAG; ++tg) acc[tg] += x * Wl[tg * 512 + H2U + u];
  }
#pragma unroll
  for (int tg = 0; tg < NTAG; ++tg) feats[(size_t)m * NTAG + tg] = acc[tg];
}

// ---------------------------------------------------------------------------
// Phase 4: Viterbi. One block, thread = (batch, tag). LDS split-barrier scan,
// then per-batch backtrace. out[0..63]=score, out[64 + b*256 + s]=path.
// ---------------------------------------------------------------------------
__global__ __launch_bounds__(640) void k_viterbi(const float* __restrict__ feats,
                                                 const float* __restrict__ trans,
                                                 int* __restrict__ bps,
                                                 float* __restrict__ out) {
  __shared__ float tr[NTAG][NTAG];
  __shared__ float fv[BATCH][NTAG + 2];
  __shared__ int   lastt[BATCH];
  int tid = threadIdx.x;
  int b = tid / NTAG, j = tid % NTAG;
  if (tid < NTAG * NTAG) tr[tid / NTAG][tid % NTAG] = trans[tid];
  fv[b][j] = (j == START_TAG) ? 0.0f : -10000.0f;
  __syncthreads();

  for (int s = 0; s < SEQ; ++s) {
    float prev[NTAG];
#pragma unroll
    for (int p = 0; p < NTAG; ++p) prev[p] = fv[b][p];
    float best = prev[0] + tr[j][0];
    int bp = 0;
#pragma unroll
    for (int p = 1; p < NTAG; ++p) {
      float v = prev[p] + tr[j][p];
      if (v > best) { best = v; bp = p; }       // strict '>' == argmax-first tie rule
    }
    float nv = best + feats[(size_t)(b * SEQ + s) * NTAG + j];
    __syncthreads();
    fv[b][j] = nv;
    bps[(size_t)(s * BATCH + b) * NTAG + j] = bp;
    __syncthreads();
  }
  // terminal scores + argmax
  float tv = fv[b][j] + tr[STOP_TAG][j];
  __syncthreads();
  fv[b][j] = tv;
  __syncthreads();
  if (j == 0) {
    float best = fv[b][0]; int bi = 0;
#pragma unroll
    for (int p = 1; p < NTAG; ++p)
      if (fv[b][p] > best) { best = fv[b][p]; bi = p; }
    out[b] = best;
    lastt[b] = bi;
  }
  __syncthreads();
  if (tid < BATCH) {
    int bb = tid, cur = lastt[bb];
    for (int s = SEQ - 1; s >= 0; --s) {
      out[BATCH + bb * SEQ + s] = (float)cur;
      cur = bps[(size_t)(s * BATCH + bb) * NTAG + cur];
    }
  }
}

// ---------------------------------------------------------------------------
extern "C" void kernel_launch(void* const* d_in, const int* in_sizes, int n_in,
                              void* d_out, int out_size, void* d_ws, size_t ws_size,
                              hipStream_t stream) {
  const int*   sent   = (const int*)  d_in[0];
  const float* emb    = (const float*)d_in[1];
  const float* wih_f  = (const float*)d_in[2];
  const float* whh_f  = (const float*)d_in[3];
  const float* bih_f  = (const float*)d_in[4];
  const float* bhh_f  = (const float*)d_in[5];
  const float* wih_b  = (const float*)d_in[6];
  const float* whh_b  = (const float*)d_in[7];
  const float* bih_b  = (const float*)d_in[8];
  const float* bhh_b  = (const float*)d_in[9];
  const float* W_out  = (const float*)d_in[10];
  const float* b_out  = (const float*)d_in[11];
  const float* h0     = (const float*)d_in[12];
  const float* c0     = (const float*)d_in[13];
  const float* transi = (const float*)d_in[14];
  float* out = (float*)d_out;

  char* ws = (char*)d_ws;
  _Float16* X16   = (_Float16*)(ws);                // 16384*320*2       = 10,485,760
  _Float16* wih16 = (_Float16*)(ws + 10485760);     // 2*1024*320*2      =  1,310,720
  _Float16* whh16 = (_Float16*)(ws + 11796480);     // 2*1024*256*2      =  1,048,576
  float*    bsum  = (float*)   (ws + 12845056);     // 2*1024*4          =      8,192
  float*    Gin   = (float*)   (ws + 12853248);     // 2*16384*1024*4    =134,217,728
  _Float16* HS    = (_Float16*)(ws + 147070976);    // 2*16384*256*2     = 16,777,216
  float*    feats = (float*)   (ws + 163848192);    // 16384*10*4        =    655,360
  int*      bps   = (int*)     (ws + 164503552);    // 256*64*10*4       =    655,360

  // Phase 0: gather + convert
  k_embed_gather<<<N_TOK, EPAD, 0, stream>>>(sent, emb, X16);
  k_cvt_pad<<<(G4 * EPAD + 255) / 256, 256, 0, stream>>>(wih_f, wih16, G4, EDIM, EPAD);
  k_cvt_pad<<<(G4 * EPAD + 255) / 256, 256, 0, stream>>>(wih_b, wih16 + (size_t)G4 * EPAD,
                                                         G4, EDIM, EPAD);
  k_cvt_pad<<<(G4 * H2U + 255) / 256, 256, 0, stream>>>(whh_f, whh16, G4, H2U, H2U);
  k_cvt_pad<<<(G4 * H2U + 255) / 256, 256, 0, stream>>>(whh_b, whh16 + (size_t)G4 * H2U,
                                                        G4, H2U, H2U);
  k_add2<<<4, 256, 0, stream>>>(bih_f, bhh_f, bsum, G4);
  k_add2<<<4, 256, 0, stream>>>(bih_b, bhh_b, bsum + G4, G4);

  // Phase 1: input projections (both directions), WMMA
  k_gin_gemm<<<4096, 256, 0, stream>>>(X16, wih16, bsum, Gin);

  // Phase 2: sequential BiLSTM recurrence (1 WGP per direction), WMMA
  k_lstm_rec<<<2, 1024, 2 * BATCH * H2U * sizeof(_Float16), stream>>>(whh16, Gin,
                                                                      h0, c0, HS);

  // Phase 3: output projection to tag space
  k_proj<<<N_TOK / 256, 256, 0, stream>>>(HS, W_out, b_out, feats);

  // Phase 4: Viterbi decode
  k_viterbi<<<1, 640, 0, stream>>>(feats, transi, bps, out);
}
